// Criss_Cross_Attention_18949395710131
// MI455X (gfx1250) — matrix operand order repaired and checked by reading, not compiled, not run on hardware
//
#include <hip/hip_runtime.h>
#include <stdint.h>

#define B_   8
#define C_   512
#define C8_  64
#define S_   128
#define NSP  16384          // S*S
#define NEG_INF_F (-1000000000.0f)

typedef __bf16 bf16;
typedef __attribute__((ext_vector_type(16))) __bf16 v16bf;
typedef __attribute__((ext_vector_type(8)))  __bf16 v8bf;
typedef __attribute__((ext_vector_type(8)))  float  v8f;

#define WMMA_BF16(acc, va, vb) \
  acc = __builtin_amdgcn_wmma_f32_16x16x32_bf16(false, va, false, vb, (short)0, acc, false, false)

// Build a 16-element bf16 fragment from two contiguous 16-byte halves.
#define FRAG16(plo, phi) \
  __builtin_shufflevector(*(const v8bf*)(plo), *(const v8bf*)(phi), \
                          0,1,2,3,4,5,6,7,8,9,10,11,12,13,14,15)

// CDNA5 async global<->LDS copies (GV mode) + ASYNCcnt waits.
#define ASYNC_LOAD_B128(ldsoff, gaddr) \
  asm volatile("global_load_async_to_lds_b128 %0, %1, off" :: "v"(ldsoff), "v"(gaddr) : "memory")
#define WAIT_ASYNCCNT_1() asm volatile("s_wait_asynccnt 1" ::: "memory")
#define WAIT_ASYNCCNT_0() asm volatile("s_wait_asynccnt 0" ::: "memory")

// ---------------------------------------------------------------------------
// Kernel 1: fused QKV projection.  QKV(640x16384) = W(640x512) * X(512x16384)
// per batch.  X tiles double-buffered in LDS via async copies.  q/k results are
// stored ONLY in K-innermost transposed layouts (one b128 store per lane each):
//   qe[b][h][w][c] , qr[b][w][h][c] , ke[b][h][u][c] , kr[b][w][u][c]
// v is stored in natural vb[b][c][w*128+u] (u-innermost, used directly by orow).
// ---------------------------------------------------------------------------
__global__ __launch_bounds__(128)
void proj_kernel(const float* __restrict__ x,
                 const float* __restrict__ Wq, const float* __restrict__ Wk,
                 const float* __restrict__ Wv,
                 const float* __restrict__ bq, const float* __restrict__ bk,
                 const float* __restrict__ bv,
                 bf16* __restrict__ qe, bf16* __restrict__ qr,
                 bf16* __restrict__ ke, bf16* __restrict__ kr,
                 bf16* __restrict__ vbn)
{
  __shared__ float xt[2][32 * 16];

  const int b    = blockIdx.z;
  const int o64  = blockIdx.y * 64;
  const int s0   = blockIdx.x * 16;   // 16 consecutive s = fixed w, h0..h0+15
  const int tid  = threadIdx.x;
  const int wave = tid >> 5;
  const int lane = tid & 31;
  const int hf   = lane >> 4;
  const int lr   = lane & 15;

  const int obase = o64 + wave * 16;
  const float* Wsrc; const float* bsrc; int ro;
  if (obase < 64)       { Wsrc = Wq; bsrc = bq; ro = obase;       }
  else if (obase < 128) { Wsrc = Wk; bsrc = bk; ro = obase - 64;  }
  else                  { Wsrc = Wv; bsrc = bv; ro = obase - 128; }

  const float*   xb   = x + (size_t)b * C_ * NSP + s0;
  const uint32_t lds0 = (uint32_t)(uintptr_t)(&xt[0][0]);

  const int lk  = tid >> 2;        // 0..31 : K row within tile
  const int lc4 = (tid & 3) * 4;   // column group
  auto issue = [&](int t, int buf) {
    uint64_t g = (uint64_t)(uintptr_t)(xb + (size_t)(t * 32 + lk) * NSP + lc4);
    uint32_t l = lds0 + (uint32_t)((buf * 512 + lk * 16 + lc4) * 4);
    ASYNC_LOAD_B128(l, g);
  };

  issue(0, 0);
  issue(1, 1);

  const float4* wrow = (const float4*)(Wsrc + (size_t)(ro + lr) * C_);

  v8f acc = {};
  for (int t = 0; t < 16; ++t) {
    if (t < 15) { WAIT_ASYNCCNT_1(); } else { WAIT_ASYNCCNT_0(); }
    __syncthreads();

    float4 f0 = wrow[t * 8 + hf * 2];
    float4 f1 = wrow[t * 8 + hf * 2 + 1];
    float4 f2 = wrow[t * 8 + 4 + hf * 2];
    float4 f3 = wrow[t * 8 + 4 + hf * 2 + 1];
    v16bf va;
    va[0]=(bf16)f0.x; va[1]=(bf16)f0.y; va[2]=(bf16)f0.z; va[3]=(bf16)f0.w;
    va[4]=(bf16)f1.x; va[5]=(bf16)f1.y; va[6]=(bf16)f1.z; va[7]=(bf16)f1.w;
    va[8]=(bf16)f2.x; va[9]=(bf16)f2.y; va[10]=(bf16)f2.z; va[11]=(bf16)f2.w;
    va[12]=(bf16)f3.x; va[13]=(bf16)f3.y; va[14]=(bf16)f3.z; va[15]=(bf16)f3.w;

    const float* xs = &xt[t & 1][0];
    v16bf vbm;
    #pragma unroll
    for (int e = 0; e < 16; ++e)
      vbm[e] = (bf16)xs[((e >> 3) * 16 + hf * 8 + (e & 7)) * 16 + lr];

    WMMA_BF16(acc, va, vbm);

    __syncthreads();
    if (t + 2 < 16) issue(t + 2, t & 1);
  }

  // D element g: (M = hf*8+g = channel offset, N = lr = h offset).
  if (obase < 128) {
    // q/k: lane owns 8 CONTIGUOUS channels -> single b128 store per layout.
    v8bf pk;
    #pragma unroll
    for (int g = 0; g < 8; ++g)
      pk[g] = (bf16)(acc[g] + bsrc[ro + hf * 8 + g]);
    const int w  = s0 >> 7;
    const int hh = (s0 & 127) + lr;
    bf16* ed = (obase < 64) ? qe : ke;
    bf16* rd = (obase < 64) ? qr : kr;
    *(v8bf*)(ed + ((size_t)(b * S_ + hh) * S_ + w) * C8_ + ro + hf * 8) = pk;
    *(v8bf*)(rd + ((size_t)(b * S_ + w) * S_ + hh) * C8_ + ro + hf * 8) = pk;
  } else {
    bf16* dst = vbn + ((size_t)b * C_ + ro) * NSP;
    #pragma unroll
    for (int g = 0; g < 8; ++g) {
      int m = hf * 8 + g;
      dst[(size_t)m * NSP + s0 + lr] = (bf16)(acc[g] + bsrc[ro + m]);
    }
  }
}

// ---------------------------------------------------------------------------
// Kernel 2: (u,h) transpose of v for the column path:
//   vcol[b][h][c][u] = v[b,c,u,h].
// Per (b,c): async-copy the 128x128 bf16 slice into LDS (source order [u][h]),
// then emit u-runs with coalesced b128 stores.
// ---------------------------------------------------------------------------
__global__ __launch_bounds__(256)
void tv_kernel(const bf16* __restrict__ vbn, bf16* __restrict__ vcol)
{
  __shared__ bf16 lt[128 * 136];          // [u][h], stride 136 keeps 16B chunks aligned
  const int b = blockIdx.y, c = blockIdx.x;
  const int tid = threadIdx.x;
  const uint32_t lds0 = (uint32_t)(uintptr_t)(&lt[0]);
  const bf16* src = vbn + (size_t)(b * C_ + c) * NSP;

  #pragma unroll
  for (int p = 0; p < 8; ++p) {
    int u  = p * 16 + (tid >> 4);
    int h8 = (tid & 15) * 8;
    uint64_t g = (uint64_t)(uintptr_t)(src + (size_t)u * S_ + h8);
    uint32_t l = lds0 + (uint32_t)((u * 136 + h8) * 2);
    ASYNC_LOAD_B128(l, g);
  }
  WAIT_ASYNCCNT_0();
  __syncthreads();

  bf16* dst = vcol + ((size_t)b * S_ * C_ + c) * S_;   // + h*512*128 later
  #pragma unroll
  for (int p = 0; p < 8; ++p) {
    int h  = p * 16 + (tid >> 4);
    int u8 = (tid & 15) * 8;
    v8bf pk;
    #pragma unroll
    for (int i = 0; i < 8; ++i)
      pk[i] = lt[(u8 + i) * 136 + h];
    *(v8bf*)(dst + (size_t)h * C_ * S_ + u8) = pk;
  }
}

// ---------------------------------------------------------------------------
// Kernel 3: column logits.  Per (b,h): E(w,u) = sum_c q[b,c,w,h] k[b,c,u,h].
// All fragments are pairs of b128 loads from qe/ke.  Diagonal masked.
// ---------------------------------------------------------------------------
__global__ __launch_bounds__(256)
void ecol_kernel(const bf16* __restrict__ qe, const bf16* __restrict__ ke,
                 float* __restrict__ e)
{
  const int b = blockIdx.y, h = blockIdx.x;
  const int wave = threadIdx.x >> 5, lane = threadIdx.x & 31;
  const int hf = lane >> 4, lr = lane & 15;
  const int w0 = wave * 16;

  const bf16* qrow = qe + ((size_t)(b * S_ + h) * S_ + (w0 + lr)) * C8_;
  v16bf va[2];
  #pragma unroll
  for (int t = 0; t < 2; ++t)
    va[t] = FRAG16(qrow + t * 32 + hf * 16, qrow + t * 32 + hf * 16 + 8);

  for (int ut = 0; ut < 8; ++ut) {
    const bf16* krow = ke + ((size_t)(b * S_ + h) * S_ + (ut * 16 + lr)) * C8_;
    v8f acc = {};
    #pragma unroll
    for (int t = 0; t < 2; ++t) {
      v16bf vbm = FRAG16(krow + t * 32 + hf * 16, krow + t * 32 + hf * 16 + 8);
      WMMA_BF16(acc, va[t], vbm);
    }
    #pragma unroll
    for (int g = 0; g < 8; ++g) {
      int w = w0 + hf * 8 + g;
      int u = ut * 16 + lr;
      e[(((size_t)b * S_ + w) * S_ + h) * 256 + u] = (w == u) ? NEG_INF_F : acc[g];
    }
  }
}

// ---------------------------------------------------------------------------
// Kernel 4: row logits.  Per (b,w): E(h,u) = sum_c q[b,c,w,h] k[b,c,w,u].
// ---------------------------------------------------------------------------
__global__ __launch_bounds__(256)
void erow_kernel(const bf16* __restrict__ qr, const bf16* __restrict__ kr,
                 float* __restrict__ e)
{
  const int b = blockIdx.y, w = blockIdx.x;
  const int wave = threadIdx.x >> 5, lane = threadIdx.x & 31;
  const int hf = lane >> 4, lr = lane & 15;
  const int h0 = wave * 16;

  const bf16* qrow = qr + ((size_t)(b * S_ + w) * S_ + (h0 + lr)) * C8_;
  v16bf va[2];
  #pragma unroll
  for (int t = 0; t < 2; ++t)
    va[t] = FRAG16(qrow + t * 32 + hf * 16, qrow + t * 32 + hf * 16 + 8);

  for (int ut = 0; ut < 8; ++ut) {
    const bf16* krow = kr + ((size_t)(b * S_ + w) * S_ + (ut * 16 + lr)) * C8_;
    v8f acc = {};
    #pragma unroll
    for (int t = 0; t < 2; ++t) {
      v16bf vbm = FRAG16(krow + t * 32 + hf * 16, krow + t * 32 + hf * 16 + 8);
      WMMA_BF16(acc, va[t], vbm);
    }
    #pragma unroll
    for (int g = 0; g < 8; ++g) {
      int h = h0 + hf * 8 + g;
      int u = ut * 16 + lr;
      e[(((size_t)b * S_ + w) * S_ + h) * 256 + 128 + u] = acc[g];
    }
  }
}

// ---------------------------------------------------------------------------
// Kernel 5: softmax over 256 logits per (b,w,h).  Outputs in [..][N][K] order
// so output-GEMM B-fragments are contiguous:
//   acolT[b][h][w][u] , arowT[b][w][h][u]   (both coalesced stores here).
// ---------------------------------------------------------------------------
__global__ __launch_bounds__(256)
void softmax_kernel(const float* __restrict__ e,
                    bf16* __restrict__ acolT, bf16* __restrict__ arowT)
{
  __shared__ float red[256];
  const int b = blockIdx.y;
  const int w = blockIdx.x >> 7;
  const int h = blockIdx.x & 127;
  const int t = threadIdx.x;
  const size_t base = (((size_t)b * S_ + w) * S_ + h) * 256;

  float v = e[base + t];
  red[t] = v; __syncthreads();
  for (int s = 128; s > 0; s >>= 1) { if (t < s) red[t] = fmaxf(red[t], red[t + s]); __syncthreads(); }
  float mx = red[0]; __syncthreads();
  float p = __expf(v - mx);
  red[t] = p; __syncthreads();
  for (int s = 128; s > 0; s >>= 1) { if (t < s) red[t] += red[t + s]; __syncthreads(); }
  float av = p / red[0];

  if (t < 128) acolT[(((size_t)b * S_ + h) * S_ + w) * S_ + t] = (bf16)av;
  else         arowT[(((size_t)b * S_ + w) * S_ + h) * S_ + (t - 128)] = (bf16)av;
}

// ---------------------------------------------------------------------------
// Kernel 6: column output.  Per (b,h): O(c,w) = sum_u v[b,c,u,h] a_col[w,h,u];
// out = x + gamma*O.  A from vcol, B from acolT — all b128 fragment loads.
// ---------------------------------------------------------------------------
__global__ __launch_bounds__(256)
void ocol_kernel(const bf16* __restrict__ vcol, const bf16* __restrict__ acolT,
                 const float* __restrict__ x, const float* __restrict__ gamma,
                 float* __restrict__ out)
{
  const int b = blockIdx.y, h = blockIdx.x;
  const int wave = threadIdx.x >> 5, lane = threadIdx.x & 31;
  const int hf = lane >> 4, lr = lane & 15;
  const int w0 = wave * 16;
  const float gm = gamma[0];

  const bf16* arow_ = acolT + ((size_t)(b * S_ + h) * S_ + (w0 + lr)) * S_;
  v16bf vbm[4];
  #pragma unroll
  for (int t = 0; t < 4; ++t)
    vbm[t] = FRAG16(arow_ + t * 32 + hf * 16, arow_ + t * 32 + hf * 16 + 8);

  for (int ct = 0; ct < 32; ++ct) {
    const bf16* vrow = vcol + ((size_t)(b * S_ + h) * C_ + (ct * 16 + lr)) * S_;
    v8f acc = {};
    #pragma unroll
    for (int t = 0; t < 4; ++t) {
      v16bf va = FRAG16(vrow + t * 32 + hf * 16, vrow + t * 32 + hf * 16 + 8);
      WMMA_BF16(acc, va, vbm[t]);
    }
    #pragma unroll
    for (int g = 0; g < 8; ++g) {
      int c = ct * 16 + hf * 8 + g;
      int w = w0 + lr;
      size_t idx = ((size_t)b * C_ + c) * NSP + (size_t)w * S_ + h;
      out[idx] = x[idx] + gm * acc[g];
    }
  }
}

// ---------------------------------------------------------------------------
// Kernel 7: row output.  Per (b,w): O(c,h) = sum_u v[b,c,w,u] a_row[w,h,u];
// out += gamma*O.  A straight from natural vb (u-innermost), B from arowT.
// ---------------------------------------------------------------------------
__global__ __launch_bounds__(256)
void orow_kernel(const bf16* __restrict__ vbn, const bf16* __restrict__ arowT,
                 const float* __restrict__ gamma, float* __restrict__ out)
{
  const int b = blockIdx.y, w = blockIdx.x;
  const int wave = threadIdx.x >> 5, lane = threadIdx.x & 31;
  const int hf = lane >> 4, lr = lane & 15;
  const int h0 = wave * 16;
  const float gm = gamma[0];

  const bf16* arow_ = arowT + ((size_t)(b * S_ + w) * S_ + (h0 + lr)) * S_;
  v16bf vbm[4];
  #pragma unroll
  for (int t = 0; t < 4; ++t)
    vbm[t] = FRAG16(arow_ + t * 32 + hf * 16, arow_ + t * 32 + hf * 16 + 8);

  for (int ct = 0; ct < 32; ++ct) {
    const bf16* vrow = vbn + ((size_t)b * C_ + ct * 16 + lr) * NSP + (size_t)w * S_;
    v8f acc = {};
    #pragma unroll
    for (int t = 0; t < 4; ++t) {
      v16bf va = FRAG16(vrow + t * 32 + hf * 16, vrow + t * 32 + hf * 16 + 8);
      WMMA_BF16(acc, va, vbm[t]);
    }
    #pragma unroll
    for (int g = 0; g < 8; ++g) {
      int c = ct * 16 + hf * 8 + g;
      int h = h0 + lr;
      size_t idx = ((size_t)b * C_ + c) * NSP + (size_t)w * S_ + h;
      out[idx] = out[idx] + gm * acc[g];
    }
  }
}

// ---------------------------------------------------------------------------
extern "C" void kernel_launch(void* const* d_in, const int* in_sizes, int n_in,
                              void* d_out, int out_size, void* d_ws, size_t ws_size,
                              hipStream_t stream) {
  const float* x  = (const float*)d_in[0];
  const float* Wq = (const float*)d_in[1];
  const float* bq = (const float*)d_in[2];
  const float* Wk = (const float*)d_in[3];
  const float* bk = (const float*)d_in[4];
  const float* Wv = (const float*)d_in[5];
  const float* bv = (const float*)d_in[6];
  const float* gm = (const float*)d_in[7];
  float* out = (float*)d_out;

  char* ws = (char*)d_ws;
  size_t off = 0;
  const size_t QSZ = (size_t)B_ * NSP * C8_ * sizeof(bf16);   // 16 MB per q/k layout
  bf16*  qe   = (bf16*)(ws + off);  off += QSZ;
  bf16*  qr   = (bf16*)(ws + off);  off += QSZ;
  bf16*  ke   = (bf16*)(ws + off);  off += QSZ;
  bf16*  kr   = (bf16*)(ws + off);  off += QSZ;
  bf16*  vbn  = (bf16*)(ws + off);  off += (size_t)B_ * C_ * NSP * sizeof(bf16);   // 128 MB
  bf16*  vcol = (bf16*)(ws + off);  off += (size_t)B_ * C_ * NSP * sizeof(bf16);   // 128 MB
  float* e    = (float*)(ws + off); off += (size_t)B_ * NSP * 256 * sizeof(float); // 128 MB
  bf16*  acolT= (bf16*)(ws + off);  off += (size_t)B_ * NSP * S_ * sizeof(bf16);   // 32 MB
  bf16*  arowT= (bf16*)(ws + off);                                                 // 32 MB

  proj_kernel   <<<dim3(NSP / 16, 10, B_), 128, 0, stream>>>(x, Wq, Wk, Wv, bq, bk, bv,
                                                             qe, qr, ke, kr, vbn);
  tv_kernel     <<<dim3(C_, B_),  256, 0, stream>>>(vbn, vcol);
  ecol_kernel   <<<dim3(S_, B_),  256, 0, stream>>>(qe, ke, e);
  erow_kernel   <<<dim3(S_, B_),  256, 0, stream>>>(qr, kr, e);
  softmax_kernel<<<dim3(NSP, B_), 256, 0, stream>>>(e, acolT, arowT);
  ocol_kernel   <<<dim3(S_, B_),  256, 0, stream>>>(vcol, acolT, x, gm, out);
  orow_kernel   <<<dim3(S_, B_),  256, 0, stream>>>(vbn, arowT, gm, out);
}
